// GCNAgg_89343909691965
// MI455X (gfx1250) — compile-verified, matrix operand order checked by reading, not run on hardware
//
#include <hip/hip_runtime.h>

// ---------------------------------------------------------------------------
// GCN aggregation layer for MI455X (gfx1250, wave32):
//   h   = x @ W^T + b                      (bf16 WMMA, f32 accumulate)
//   agg = segment_sum(edge_vals * h[cols]) (edge-parallel, f32 L2 atomics)
//   out = leaky_relu(agg, 0.01)
//
// h is stored bf16 (51.2 MB) so the 3.2M random row-gathers of the SpMM stay
// resident in the 192 MB L2 and gather traffic is halved vs f32.
//
// GEMM: each wave owns a 16x64 strip -> one A fragment feeds 4 WMMAs per
// K-step (A-operand reuse), and a block's 8 waves cover all 256 output cols,
// so x (the 102 MB operand) is fetched only 4x total instead of 16x.
// ---------------------------------------------------------------------------

typedef __attribute__((ext_vector_type(16))) __bf16 v16bf;
typedef __attribute__((ext_vector_type(8)))  float  v8f;

#define DDIM 256

// ---- W (f32 DxD) -> bf16 ---------------------------------------------------
__global__ __launch_bounds__(256)
void cvt_w_kernel(const float* __restrict__ W, __bf16* __restrict__ Wb, int n) {
    int i = blockIdx.x * 256 + threadIdx.x;
    if (i < n) Wb[i] = (__bf16)W[i];
}

// ---- GEMM: h[n, e] = sum_d x[n,d] * W[e,d] + b[e]   (bf16 WMMA) ------------
// Block = 256 threads = 8 waves. Wave w: M half = w>>2 (16 rows each),
// N quad = w&3 (64 cols each) -> block tile = 32 x 256 (full N).
// Per K-step: 1 A fragment, 4 B fragments, 4 chained v_wmma_f32_16x16x32_bf16.
__global__ __launch_bounds__(256)
void gemm_bf16_wmma_kernel(const float* __restrict__ x,
                           const __bf16* __restrict__ Wb,
                           const float* __restrict__ bias,
                           __bf16* __restrict__ h,
                           int n_nodes) {
    const int lane = threadIdx.x & 31;
    const int wave = threadIdx.x >> 5;
    const int lr   = lane & 15;        // row (A) / col (B) within a 16-wide tile
    const int half = lane >> 4;        // K-interleave half (ISA 16-bit layout)
    const int koff = half * 8;         // lanes 0-15: K {0..7,16..23}; 16-31: {8..15,24..31}

    const int m_tile = blockIdx.x * 32 + (wave >> 2) * 16;
    const int n_base = (wave & 3) * 64;           // 4 consecutive 16-col tiles

    int arow = m_tile + lr;
    if (arow >= n_nodes) arow = n_nodes - 1;      // clamp: keep EXEC all-ones for WMMA
    const float* xrow = x + (size_t)arow * DDIM;

    // B[k,n] = W[n,k] -> lane lr of tile t reads row (n_base + 16t + lr) of W.
    const __bf16* wrow0 = Wb + (size_t)(n_base +  0 + lr) * DDIM;
    const __bf16* wrow1 = Wb + (size_t)(n_base + 16 + lr) * DDIM;
    const __bf16* wrow2 = Wb + (size_t)(n_base + 32 + lr) * DDIM;
    const __bf16* wrow3 = Wb + (size_t)(n_base + 48 + lr) * DDIM;

    v8f acc0 = {}, acc1 = {}, acc2 = {}, acc3 = {};

    #pragma unroll
    for (int k = 0; k < DDIM; k += 32) {
        // A fragment: convert 16 fp32 -> bf16 in ISA lane order, reused 4x.
        v16bf va;
        const float* xp = xrow + k + koff;
        #pragma unroll
        for (int j = 0; j < 8; ++j) {
            va[j]     = (__bf16)xp[j];         // K = k+koff+j
            va[j + 8] = (__bf16)xp[16 + j];    // K = k+16+koff+j
        }

        v16bf vb0, vb1, vb2, vb3;
        #pragma unroll
        for (int j = 0; j < 8; ++j) {
            vb0[j] = wrow0[k + koff + j];  vb0[j + 8] = wrow0[k + 16 + koff + j];
            vb1[j] = wrow1[k + koff + j];  vb1[j + 8] = wrow1[k + 16 + koff + j];
            vb2[j] = wrow2[k + koff + j];  vb2[j + 8] = wrow2[k + 16 + koff + j];
            vb3[j] = wrow3[k + koff + j];  vb3[j + 8] = wrow3[k + 16 + koff + j];
        }

        acc0 = __builtin_amdgcn_wmma_f32_16x16x32_bf16(false, va, false, vb0,
                                                       (short)0, acc0, false, false);
        acc1 = __builtin_amdgcn_wmma_f32_16x16x32_bf16(false, va, false, vb1,
                                                       (short)0, acc1, false, false);
        acc2 = __builtin_amdgcn_wmma_f32_16x16x32_bf16(false, va, false, vb2,
                                                       (short)0, acc2, false, false);
        acc3 = __builtin_amdgcn_wmma_f32_16x16x32_bf16(false, va, false, vb3,
                                                       (short)0, acc3, false, false);
    }

    // C/D layout: VGPR r, lanes 0-15 -> (M=r, N=lane); lanes 16-31 -> (M=8+r).
    v8f accs[4] = {acc0, acc1, acc2, acc3};
    #pragma unroll
    for (int t = 0; t < 4; ++t) {
        const int   col  = n_base + t * 16 + lr;
        const float bcol = bias[col];
        #pragma unroll
        for (int r = 0; r < 8; ++r) {
            int row = m_tile + half * 8 + r;
            if (row < n_nodes)
                h[(size_t)row * DDIM + col] = (__bf16)(accs[t][r] + bcol);
        }
    }
}

// ---- zero d_out (harness poisons it) ---------------------------------------
__global__ __launch_bounds__(256)
void zero_kernel(float4* __restrict__ out, long n4) {
    long i = (long)blockIdx.x * 256 + threadIdx.x;
    if (i < n4) out[i] = make_float4(0.f, 0.f, 0.f, 0.f);
}

// ---- SpMM: out[r] += v_e * h[c_e]   (one wave per edge, 8 cols per lane) ---
__global__ __launch_bounds__(256)
void spmm_kernel(const int*   __restrict__ erows,
                 const int*   __restrict__ ecols,
                 const float* __restrict__ evals,
                 const __bf16* __restrict__ h,
                 float* __restrict__ out,
                 int n_edges) {
    const int lane = threadIdx.x & 31;
    const int wave = threadIdx.x >> 5;
    const int e = blockIdx.x * 8 + wave;
    if (e >= n_edges) return;

    const int   c = ecols[e];
    const int   r = erows[e];
    const float v = evals[e];

    // 8 bf16 = 16 bytes per lane -> single global_load_b128, L2-resident.
    const __bf16* hp = h   + (size_t)c * DDIM + lane * 8;
    float*        op = out + (size_t)r * DDIM + lane * 8;
    #pragma unroll
    for (int j = 0; j < 8; ++j) {
        atomicAdd(op + j, v * (float)hp[j]);   // global_atomic_add_f32 (no return)
    }
}

// ---- leaky_relu in place ---------------------------------------------------
__global__ __launch_bounds__(256)
void lrelu_kernel(float4* __restrict__ out, long n4) {
    long i = (long)blockIdx.x * 256 + threadIdx.x;
    if (i < n4) {
        float4 a = out[i];
        a.x = a.x >= 0.f ? a.x : 0.01f * a.x;
        a.y = a.y >= 0.f ? a.y : 0.01f * a.y;
        a.z = a.z >= 0.f ? a.z : 0.01f * a.z;
        a.w = a.w >= 0.f ? a.w : 0.01f * a.w;
        out[i] = a;
    }
}

extern "C" void kernel_launch(void* const* d_in, const int* in_sizes, int n_in,
                              void* d_out, int out_size, void* d_ws, size_t ws_size,
                              hipStream_t stream) {
    const float* x     = (const float*)d_in[0];
    const int*   erows = (const int*)  d_in[1];
    const int*   ecols = (const int*)  d_in[2];
    const float* evals = (const float*)d_in[3];
    const float* W_w   = (const float*)d_in[4];
    const float* W_b   = (const float*)d_in[5];
    float*       out   = (float*)d_out;

    const int n_nodes = in_sizes[0] / DDIM;
    const int n_edges = in_sizes[1];

    // Workspace layout: [ h : n_nodes*DDIM bf16 ][ W_bf16 : DDIM*DDIM bf16 ]
    __bf16* h   = (__bf16*)d_ws;
    __bf16* Wbf = (__bf16*)((char*)d_ws + (size_t)n_nodes * DDIM * sizeof(__bf16));

    // 1) W -> bf16
    {
        int n = DDIM * DDIM;
        cvt_w_kernel<<<(n + 255) / 256, 256, 0, stream>>>(W_w, Wbf, n);
    }
    // 2) h = x @ W^T + b   (bf16 WMMA GEMM; one block covers 32 rows x all 256 cols)
    {
        dim3 grid((n_nodes + 31) / 32);
        gemm_bf16_wmma_kernel<<<grid, 256, 0, stream>>>(x, Wbf, W_b, h, n_nodes);
    }
    // 3) zero the output accumulator
    {
        long n4 = (long)n_nodes * DDIM / 4;
        zero_kernel<<<(unsigned)((n4 + 255) / 256), 256, 0, stream>>>((float4*)out, n4);
    }
    // 4) edge-parallel SpMM with f32 atomics
    {
        int blocks = (n_edges + 7) / 8;
        spmm_kernel<<<blocks, 256, 0, stream>>>(erows, ecols, evals, h, out, n_edges);
    }
    // 5) leaky_relu in place
    {
        long n4 = (long)n_nodes * DDIM / 4;
        lrelu_kernel<<<(unsigned)((n4 + 255) / 256), 256, 0, stream>>>((float4*)out, n4);
    }
}
